// FusedMultiHeadAttention_86199993631157
// MI455X (gfx1250) — matrix operand order repaired and checked by reading, not compiled, not run on hardware
//
#include <hip/hip_runtime.h>
#include <hip/hip_bf16.h>
#include <stdint.h>
#include <stddef.h>

// Problem constants (match reference)
constexpr int Bc = 2, Sc = 2048, Dc = 1024, Hc = 16, DKc = 64;
constexpr int BSc = Bc * Sc;
constexpr float SCALEc = 0.125f; // 1/sqrt(64)

typedef __attribute__((ext_vector_type(16))) __bf16 bf16x16;
typedef __attribute__((ext_vector_type(8)))  float  f32x8;
typedef __attribute__((ext_vector_type(4)))  int    v4i;

// gfx1250 async Memory->LDS path (ASYNCcnt), guarded so compile never breaks.
#if __has_builtin(__builtin_amdgcn_global_load_async_to_lds_b128)
#define ASYNC_LDS 1
#else
#define ASYNC_LDS 0
#endif

__device__ __forceinline__ void ldsStage16(const __bf16* g, __bf16* l) {
#if ASYNC_LDS
  // Signature (from hipcc diagnostic): (v4i AS1* src, v4i AS3* dst, imm off, imm cpol)
  __builtin_amdgcn_global_load_async_to_lds_b128(
      (__attribute__((address_space(1))) v4i*)g,
      (__attribute__((address_space(3))) v4i*)l, 0, 0);
#else
  *(uint4*)l = *(const uint4*)g;
#endif
}

__device__ __forceinline__ void ldsStageWait() {
#if ASYNC_LDS
#if __has_builtin(__builtin_amdgcn_s_wait_asynccnt)
  __builtin_amdgcn_s_wait_asynccnt(0);
#else
  asm volatile("s_wait_asynccnt 0" ::: "memory");
#endif
#endif
}

__device__ __forceinline__ __bf16 f2bf(float f) {
  union { float f; uint32_t u; } v; v.f = f;
  uint32_t r = v.u + 0x7FFFu + ((v.u >> 16) & 1u);  // round-to-nearest-even
  uint16_t h = (uint16_t)(r >> 16);
  __bf16 o; __builtin_memcpy(&o, &h, 2);
  return o;
}

__device__ __forceinline__ f32x8 wmma_bf16(bf16x16 a, bf16x16 b, f32x8 c) {
  // (neg_a, A, neg_b, B, c_mod, C, reuse_a, reuse_b)
  return __builtin_amdgcn_wmma_f32_16x16x32_bf16(false, a, false, b, (short)0, c,
                                                 false, false);
}

// ---------------------------------------------------------------------------
// Kernel: f32 -> bf16 conversion (grid-stride)
// ---------------------------------------------------------------------------
__global__ void cvt_f32_bf16(const float* __restrict__ src,
                             __bf16* __restrict__ dst, int n) {
  int i = blockIdx.x * blockDim.x + threadIdx.x;
  int stride = gridDim.x * blockDim.x;
  for (; i < n; i += stride) dst[i] = f2bf(src[i]);
}

// ---------------------------------------------------------------------------
// Shared GEMM mainloop: C_tile(64x128) = A[m0:m0+64, :K] * W[n0:n0+128, :K]^T
// A row-major [M,K], W row-major [N,K] (i.e. computes A @ W^T).
// 256 threads = 8 waves in 2x4; each wave: 2x2 WMMA 16x16 tiles, K-step 32.
// Double-buffered LDS tiles: async fill of tile t+1 overlaps WMMA on tile t.
// ---------------------------------------------------------------------------
__device__ __forceinline__ void stage_gemm_tiles(
    const __bf16* __restrict__ A, const __bf16* __restrict__ W,
    int m0, int n0, int K, int k0,
    __bf16 (&As)[64][40], __bf16 (&Bs)[128][40]) {
  const int tid = threadIdx.x;
  const int arow = tid >> 2, aseg = tid & 3;
  ldsStage16(A + (size_t)(m0 + arow) * (size_t)K + k0 + aseg * 8,
             &As[arow][aseg * 8]);
#pragma unroll
  for (int rep = 0; rep < 2; ++rep) {
    const int brow = arow + rep * 64;
    ldsStage16(W + (size_t)(n0 + brow) * (size_t)K + k0 + aseg * 8,
               &Bs[brow][aseg * 8]);
  }
}

__device__ __forceinline__ void gemm_mainloop_64x128(
    const __bf16* __restrict__ A, const __bf16* __restrict__ W,
    int m0, int n0, int K,
    __bf16 (&As)[2][64][40], __bf16 (&Bs)[2][128][40], f32x8 (&acc)[2][2]) {
  const int tid  = threadIdx.x;
  const int wave = tid >> 5, lane = tid & 31;
  const int wm = wave >> 2, wn = wave & 3;
  const int r = lane & 15, hi = lane >> 4;
  const int koffA = hi * 8;    // A-frag K sub-offset per half-wave
  const int kbase = hi * 16;   // unused; B-frag is indexed with koffA, like A
  const int arow = tid >> 2, aseg = tid & 3;

  stage_gemm_tiles(A, W, m0, n0, K, 0, As[0], Bs[0]);
  ldsStageWait();
  __syncthreads();

  for (int k0 = 0; k0 < K; k0 += 32) {
    const int cur = (k0 >> 5) & 1;
    if (k0 + 32 < K)  // async engine fills the other buffer while we compute
      stage_gemm_tiles(A, W, m0, n0, K, k0 + 32, As[cur ^ 1], Bs[cur ^ 1]);
    if (k0 + 64 < K) {  // emits global_prefetch_b8 (gfx1250)
      __builtin_prefetch(A + (size_t)(m0 + arow) * (size_t)K + k0 + 64 + aseg * 8, 0, 1);
      __builtin_prefetch(W + (size_t)(n0 + arow) * (size_t)K + k0 + 64 + aseg * 8, 0, 1);
    }

    // A fragment: lanes 0-15 hold row r with K = {0..7,16..23}; lanes 16-31
    // hold row r with K = {8..15,24..31} (ISA 7.12.2, 16-bit A 16x32).
    bf16x16 af[2];
#pragma unroll
    for (int i = 0; i < 2; ++i) {
      const __bf16* ap = &As[cur][wm * 32 + i * 16 + r][0];
#pragma unroll
      for (int e = 0; e < 8; ++e) af[i][e] = ap[koffA + e];
#pragma unroll
      for (int e = 0; e < 8; ++e) af[i][8 + e] = ap[16 + koffA + e];
    }
    // B fragment: lane = column, same K order as A ({0..7,16..23} / {8..15,24..31}).
    bf16x16 bfr[2];
#pragma unroll
    for (int j = 0; j < 2; ++j) {
      const int ncol = wn * 32 + j * 16 + r;
#pragma unroll
      for (int e = 0; e < 16; ++e) bfr[j][e] = Bs[cur][ncol][koffA + e + (e >= 8 ? 8 : 0)];
    }
#pragma unroll
    for (int i = 0; i < 2; ++i)
#pragma unroll
      for (int j = 0; j < 2; ++j) acc[i][j] = wmma_bf16(af[i], bfr[j], acc[i][j]);

    ldsStageWait();   // next tile resident in LDS
    __syncthreads();  // all waves done reading current tile
  }
}

// ---------------------------------------------------------------------------
// Kernel: QKV projection. C[BS,3D] = x_bf @ qkv_w_bf^T + qkv_b, scattered into
// q/k/v buffers laid out [B,H,S,DK] in bf16.
// ---------------------------------------------------------------------------
__global__ __launch_bounds__(256) void qkv_gemm(
    const __bf16* __restrict__ A, const __bf16* __restrict__ W,
    const float* __restrict__ bias, __bf16* __restrict__ qb,
    __bf16* __restrict__ kb, __bf16* __restrict__ vb) {
  __shared__ __align__(16) __bf16 As[2][64][40];
  __shared__ __align__(16) __bf16 Bs[2][128][40];
  const int tid = threadIdx.x;
  const int wave = tid >> 5, lane = tid & 31;
  const int wm = wave >> 2, wn = wave & 3;
  const int r = lane & 15, hi = lane >> 4;
  const int m0 = blockIdx.x * 64;
  const int n0 = blockIdx.y * 128;

  f32x8 acc[2][2] = {};
  gemm_mainloop_64x128(A, W, m0, n0, Dc, As, Bs, acc);

#pragma unroll
  for (int i = 0; i < 2; ++i)
#pragma unroll
    for (int j = 0; j < 2; ++j) {
      const int n = n0 + wn * 32 + j * 16 + r;   // column (lane)
      const float bv = bias[n];
      const int which = n >> 10;                 // 0=q 1=k 2=v
      const int d = n & 1023, h = d >> 6, dk = d & 63;
      __bf16* dst = (which == 0) ? qb : ((which == 1) ? kb : vb);
#pragma unroll
      for (int vi = 0; vi < 8; ++vi) {
        const int m = m0 + wm * 32 + i * 16 + vi + hi * 8;  // row
        const int bidx = m >> 11, s = m & 2047;
        dst[(size_t)((bidx * Hc + h) * Sc + s) * DKc + dk] =
            f2bf(acc[i][j][vi] + bv);
      }
    }
}

// ---------------------------------------------------------------------------
// Kernel: flash attention. grid = (S/128, B*H); 8 waves, wave owns 16 q rows.
// Double-buffered K/V tiles staged through the async LDS path.
// ---------------------------------------------------------------------------
__device__ __forceinline__ void stage_kv(
    const __bf16* __restrict__ kptr, const __bf16* __restrict__ vptr, int kb,
    __bf16 (&Kt)[32][72], __bf16 (&Vt)[32][72]) {
  const int tid = threadIdx.x;
  const int ldrow = tid >> 3, ldseg = tid & 7;
  ldsStage16(kptr + (size_t)(kb * 32 + ldrow) * DKc + ldseg * 8,
             &Kt[ldrow][ldseg * 8]);
  ldsStage16(vptr + (size_t)(kb * 32 + ldrow) * DKc + ldseg * 8,
             &Vt[ldrow][ldseg * 8]);
}

__global__ __launch_bounds__(256) void flash_attn(
    const __bf16* __restrict__ qg, const __bf16* __restrict__ kg,
    const __bf16* __restrict__ vg, const int* __restrict__ mask,
    __bf16* __restrict__ og) {
  __shared__ __align__(16) __bf16 Kt[2][32][72];   // [buf][key][dk], padded
  __shared__ __align__(16) __bf16 Vt[2][32][72];
  __shared__ __align__(16) __bf16 Pw[8][16][40];   // per-wave P transpose

  const int tid = threadIdx.x;
  const int wave = tid >> 5, lane = tid & 31;
  const int r = lane & 15, hi = lane >> 4;
  const int koff = hi * 8, kbase = hi * 16;
  const int bh = blockIdx.y;
  const int bidx = bh / Hc, h = bh % Hc;
  const int q0 = blockIdx.x * 128 + wave * 16;
  const size_t head_off = (size_t)(bidx * Hc + h) * Sc * DKc;
  const __bf16* qbase = qg + head_off;
  const __bf16* kptr  = kg + head_off;
  const __bf16* vptr  = vg + head_off;

  // Q fragments for DK=64 split into two K=32 halves (held in registers).
  bf16x16 qf[2];
  {
    const __bf16* qrow = qbase + (size_t)(q0 + r) * DKc;
#pragma unroll
    for (int ko = 0; ko < 2; ++ko) {
#pragma unroll
      for (int e = 0; e < 8; ++e) qf[ko][e] = qrow[ko * 32 + koff + e];
#pragma unroll
      for (int e = 0; e < 8; ++e) qf[ko][8 + e] = qrow[ko * 32 + 16 + koff + e];
    }
  }

  float mrow[8], lrow[8];
#pragma unroll
  for (int vi = 0; vi < 8; ++vi) { mrow[vi] = -3.0e38f; lrow[vi] = 0.0f; }
  f32x8 oacc[4] = {};

  constexpr int NB = Sc / 32;
  stage_kv(kptr, vptr, 0, Kt[0], Vt[0]);
  ldsStageWait();
  __syncthreads();

  for (int kb = 0; kb < NB; ++kb) {
    const int cur = kb & 1;
    if (kb + 1 < NB)  // async fill of next K/V tile overlaps the math below
      stage_kv(kptr, vptr, kb + 1, Kt[cur ^ 1], Vt[cur ^ 1]);

    // S = Q (16x64) @ K^T (64x32): two 16-key column tiles, two K=32 halves.
    f32x8 sacc[2] = {};
#pragma unroll
    for (int j = 0; j < 2; ++j) {
      const int keyc = j * 16 + r;
#pragma unroll
      for (int ko = 0; ko < 2; ++ko) {
        bf16x16 bfr;
#pragma unroll
        for (int e = 0; e < 16; ++e) bfr[e] = Kt[cur][keyc][ko * 32 + koff + e + (e >= 8 ? 8 : 0)];
        sacc[j] = wmma_bf16(qf[ko], bfr, sacc[j]);
      }
    }

    // Scale + mask (faithful to reference; mask is L2-resident).
#pragma unroll
    for (int j = 0; j < 2; ++j) {
      const int key = kb * 32 + j * 16 + r;
#pragma unroll
      for (int vi = 0; vi < 8; ++vi) {
        const int qr = q0 + vi + hi * 8;
        float s = sacc[j][vi] * SCALEc;
        if (mask[(size_t)(bidx * Sc + qr) * Sc + key] == 0) s = -1.0e9f;
        sacc[j][vi] = s;
      }
    }

    // Online softmax: rows live across the 16 lanes of each half-wave.
#pragma unroll
    for (int vi = 0; vi < 8; ++vi) {
      float v = fmaxf(sacc[0][vi], sacc[1][vi]);
#pragma unroll
      for (int off = 1; off < 16; off <<= 1) v = fmaxf(v, __shfl_xor(v, off, 32));
      const float mnew = fmaxf(mrow[vi], v);
      const float alpha = __expf(mrow[vi] - mnew);
      const float p0 = __expf(sacc[0][vi] - mnew);
      const float p1 = __expf(sacc[1][vi] - mnew);
      sacc[0][vi] = p0; sacc[1][vi] = p1;
      float rs = p0 + p1;
#pragma unroll
      for (int off = 1; off < 16; off <<= 1) rs += __shfl_xor(rs, off, 32);
      lrow[vi] = lrow[vi] * alpha + rs;
      mrow[vi] = mnew;
#pragma unroll
      for (int t = 0; t < 4; ++t) oacc[t][vi] *= alpha;
    }

    // Transpose P (C-layout -> A-layout) through per-wave LDS scratch.
#pragma unroll
    for (int j = 0; j < 2; ++j)
#pragma unroll
      for (int vi = 0; vi < 8; ++vi)
        Pw[wave][vi + hi * 8][j * 16 + r] = f2bf(sacc[j][vi]);

    bf16x16 pf;
#pragma unroll
    for (int e = 0; e < 8; ++e) pf[e] = Pw[wave][r][koff + e];
#pragma unroll
    for (int e = 0; e < 8; ++e) pf[8 + e] = Pw[wave][r][16 + koff + e];

    // O += P (16x32) @ V (32x64): four 16-col output tiles.
#pragma unroll
    for (int t = 0; t < 4; ++t) {
      bf16x16 bfr;
#pragma unroll
      for (int e = 0; e < 16; ++e) bfr[e] = Vt[cur][koff + e + (e >= 8 ? 8 : 0)][t * 16 + r];
      oacc[t] = wmma_bf16(pf, bfr, oacc[t]);
    }

    ldsStageWait();   // next K/V tile resident
    __syncthreads();  // all waves done reading current tile
  }

  // Normalize and write [B,S,D] bf16 (D index = h*64 + dk).
#pragma unroll
  for (int t = 0; t < 4; ++t)
#pragma unroll
    for (int vi = 0; vi < 8; ++vi) {
      const int s = q0 + vi + hi * 8;
      const int dk = t * 16 + r;
      og[(size_t)(bidx * Sc + s) * Dc + h * DKc + dk] =
          f2bf(oacc[t][vi] / lrow[vi]);
    }
}

// ---------------------------------------------------------------------------
// Kernel: output projection. out[BS,D] = attn_bf @ out_w_bf^T + out_b (f32).
// ---------------------------------------------------------------------------
__global__ __launch_bounds__(256) void out_gemm(
    const __bf16* __restrict__ A, const __bf16* __restrict__ W,
    const float* __restrict__ bias, float* __restrict__ out) {
  __shared__ __align__(16) __bf16 As[2][64][40];
  __shared__ __align__(16) __bf16 Bs[2][128][40];
  const int tid = threadIdx.x;
  const int wave = tid >> 5, lane = tid & 31;
  const int wm = wave >> 2, wn = wave & 3;
  const int r = lane & 15, hi = lane >> 4;
  const int m0 = blockIdx.x * 64;
  const int n0 = blockIdx.y * 128;

  f32x8 acc[2][2] = {};
  gemm_mainloop_64x128(A, W, m0, n0, Dc, As, Bs, acc);

#pragma unroll
  for (int i = 0; i < 2; ++i)
#pragma unroll
    for (int j = 0; j < 2; ++j) {
      const int n = n0 + wn * 32 + j * 16 + r;
      const float bv = bias[n];
#pragma unroll
      for (int vi = 0; vi < 8; ++vi) {
        const int m = m0 + wm * 32 + i * 16 + vi + hi * 8;
        out[(size_t)m * Dc + n] = acc[i][j][vi] + bv;
      }
    }
}

// ---------------------------------------------------------------------------
// Launch
// ---------------------------------------------------------------------------
extern "C" void kernel_launch(void* const* d_in, const int* in_sizes, int n_in,
                              void* d_out, int out_size, void* d_ws,
                              size_t ws_size, hipStream_t stream) {
  const float* x     = (const float*)d_in[0];
  const int*   mask  = (const int*)d_in[1];
  const float* qkv_w = (const float*)d_in[2];
  const float* qkv_b = (const float*)d_in[3];
  const float* out_w = (const float*)d_in[4];
  const float* out_b = (const float*)d_in[5];
  float* out = (float*)d_out;

  // Workspace carve-up (bf16 staging buffers, ~48 MB total, 256B aligned).
  char* ws = (char*)d_ws;
  size_t off = 0;
  auto carve = [&](size_t elems) -> __bf16* {
    __bf16* p = (__bf16*)(ws + off);
    off += (elems * sizeof(__bf16) + 255) & ~(size_t)255;
    return p;
  };
  __bf16* x_bf    = carve((size_t)BSc * Dc);        // [BS, D]
  __bf16* wqkv_bf = carve((size_t)3 * Dc * Dc);     // [3D, D]
  __bf16* wout_bf = carve((size_t)Dc * Dc);         // [D, D]
  __bf16* qbuf    = carve((size_t)Bc * Hc * Sc * DKc);
  __bf16* kbuf    = carve((size_t)Bc * Hc * Sc * DKc);
  __bf16* vbuf    = carve((size_t)Bc * Hc * Sc * DKc);
  __bf16* attn_bf = carve((size_t)BSc * Dc);        // [BS, D]
  (void)ws_size; (void)in_sizes; (void)n_in; (void)out_size;

  // 1) precision conversion
  cvt_f32_bf16<<<1024, 256, 0, stream>>>(x, x_bf, BSc * Dc);
  cvt_f32_bf16<<<1024, 256, 0, stream>>>(qkv_w, wqkv_bf, 3 * Dc * Dc);
  cvt_f32_bf16<<<512, 256, 0, stream>>>(out_w, wout_bf, Dc * Dc);

  // 2) QKV projection
  qkv_gemm<<<dim3(BSc / 64, (3 * Dc) / 128), 256, 0, stream>>>(
      x_bf, wqkv_bf, qkv_b, qbuf, kbuf, vbuf);

  // 3) flash attention
  flash_attn<<<dim3(Sc / 128, Bc * Hc), 256, 0, stream>>>(
      qbuf, kbuf, vbuf, mask, attn_bf);

  // 4) output projection
  out_gemm<<<dim3(BSc / 64, Dc / 128), 256, 0, stream>>>(
      attn_bf, wout_bf, out_b, out);
}
